// DeepSeekMoEFFN_62474594288229
// MI455X (gfx1250) — matrix operand order repaired and checked by reading, not compiled, not run on hardware
//
#include <hip/hip_runtime.h>
#include <hip/hip_bf16.h>

typedef __attribute__((ext_vector_type(16))) __bf16 bf16x16;
typedef __attribute__((ext_vector_type(8)))  __bf16 bf16x8;
typedef __attribute__((ext_vector_type(4)))  __bf16 bf16x4;
typedef __attribute__((ext_vector_type(8)))  float  v8f;

#define D_MODEL 1024
#define HIDDEN  4096
#define NTOK    16384

#define BM 128
#define BN 128
#define BK 32
#define LDT 40   // padded LDS row stride in bf16 elems (stride 20 banks -> conflict free over 16 lanes)

__device__ __forceinline__ bf16x16 cat16(bf16x8 lo, bf16x8 hi) {
  return __builtin_shufflevector(lo, hi, 0,1,2,3,4,5,6,7,8,9,10,11,12,13,14,15);
}

// ---------------- prep kernels ----------------

__global__ void k_init(int* meta) {
  if (threadIdx.x == 0) { meta[0] = 0; meta[1] = 0; }
}

// gate scores + two-ended routing permutation (expert0 from front, expert1 from back)
__global__ void k_gate(const float* __restrict__ x, const float* __restrict__ gw,
                       int* __restrict__ meta, int* __restrict__ rowmap) {
  int wave = threadIdx.x >> 5;
  int lane = threadIdx.x & 31;
  int t = blockIdx.x * 8 + wave;
  const float* xr = x + (size_t)t * D_MODEL;
  float s0 = 0.f, s1 = 0.f;
  for (int k = lane; k < D_MODEL; k += 32) {
    float xv = xr[k];
    s0 += xv * gw[k];
    s1 += xv * gw[D_MODEL + k];
  }
  for (int off = 16; off; off >>= 1) {
    s0 += __shfl_down(s0, off, 32);
    s1 += __shfl_down(s1, off, 32);
  }
  if (lane == 0) {
    int e = (s1 > s0) ? 1 : 0;   // argmax picks first index on tie
    int slot = (e == 0) ? atomicAdd(&meta[0], 1) : (NTOK - 1 - atomicAdd(&meta[1], 1));
    rowmap[slot] = t;
  }
}

// f32 [K][Nout] -> bf16 [Nout][K]  (tiled transpose+convert)
__global__ void k_tconv(const float* __restrict__ src, __bf16* __restrict__ dst,
                        int K, int Nout) {
  __shared__ float tile[32][33];
  int n  = blockIdx.x * 32 + threadIdx.x;
  int k0 = blockIdx.y * 32;
  #pragma unroll
  for (int j = 0; j < 32; j += 8)
    tile[threadIdx.y + j][threadIdx.x] = src[(size_t)(k0 + threadIdx.y + j) * Nout + n];
  __syncthreads();
  int nn = blockIdx.x * 32;
  #pragma unroll
  for (int j = 0; j < 32; j += 8)
    dst[(size_t)(nn + threadIdx.y + j) * K + k0 + threadIdx.x] =
        (__bf16)tile[threadIdx.x][threadIdx.y + j];
}

// cAdd[e] = relu(b1[1-e]) @ w2[1-e] + b2[1-e]  (constant leaked into tokens routed to e)
__global__ void k_const(const float* __restrict__ b1, const float* __restrict__ w2,
                        const float* __restrict__ b2, float* __restrict__ cAdd) {
  int e = blockIdx.y;
  int d = blockIdx.x * 256 + threadIdx.x;
  const float* w = w2 + (size_t)e * HIDDEN * D_MODEL + d;
  const float* b = b1 + (size_t)e * HIDDEN;
  float s = 0.f;
  for (int h = 0; h < HIDDEN; ++h) s += fmaxf(b[h], 0.f) * w[(size_t)h * D_MODEL];
  cAdd[(size_t)(1 - e) * D_MODEL + d] = s + b2[(size_t)e * D_MODEL + d];
}

// ---------------- GEMM layer 1: h = relu(x_gathered @ W1 + b1), bf16 out ----------------

__global__ __launch_bounds__(256) void k_gemm1(
    const float* __restrict__ x, const int* __restrict__ rowmap,
    const int* __restrict__ meta, const __bf16* __restrict__ BtBase,
    const float* __restrict__ biasBase, __bf16* __restrict__ hbuf, int mode) {
  constexpr int K = D_MODEL, NO = HIDDEN;
  __shared__ __bf16 As[BM][LDT];
  __shared__ __bf16 Bs[BN][LDT];
  __shared__ int rowTok[BM];

  int r0 = blockIdx.y * BM;
  int n0 = blockIdx.x * BN;
  int e = (mode == 1) ? (int)blockIdx.z : 0;
  int segLo = 0, segHi = NTOK;
  if (mode == 1) { int c0 = meta[0]; segLo = e ? c0 : 0; segHi = e ? NTOK : c0; }
  if (r0 >= segHi || r0 + BM <= segLo) return;

  const __bf16* Bt  = BtBase + (size_t)e * NO * K;
  const float* bias = biasBase + (size_t)e * NO;

  int tid = threadIdx.x, lane = tid & 31, wave = tid >> 5;
  int wm0 = (wave & 3) * 32, wn0 = (wave >> 2) * 64;
  int lh = lane >> 4, l15 = lane & 15;

  if (tid < BM) rowTok[tid] = rowmap[r0 + tid];
  __syncthreads();

  int lr = tid >> 3;          // 0..31
  int lc = (tid & 7) * 4;     // 0..28

  v8f acc[2][4];
  v8f vzero = {0.f,0.f,0.f,0.f,0.f,0.f,0.f,0.f};
  #pragma unroll
  for (int mi = 0; mi < 2; ++mi)
    #pragma unroll
    for (int ni = 0; ni < 4; ++ni) acc[mi][ni] = vzero;

  for (int kt = 0; kt < K; kt += BK) {
    float4 av[4];
    bf16x4 bv[4];
    #pragma unroll
    for (int i = 0; i < 4; ++i) {
      int r = lr + i * 32;
      int tok = rowTok[r];
      av[i] = *(const float4*)(x + (size_t)tok * K + kt + lc);
      bv[i] = *(const bf16x4*)(Bt + (size_t)(n0 + r) * K + kt + lc);
    }
    __syncthreads();
    #pragma unroll
    for (int i = 0; i < 4; ++i) {
      int r = lr + i * 32;
      bf16x4 p;
      p[0] = (__bf16)av[i].x; p[1] = (__bf16)av[i].y;
      p[2] = (__bf16)av[i].z; p[3] = (__bf16)av[i].w;
      *(bf16x4*)&As[r][lc] = p;
      *(bf16x4*)&Bs[r][lc] = bv[i];
    }
    __syncthreads();

    bf16x16 af[2], bfrag[4];
    #pragma unroll
    for (int mi = 0; mi < 2; ++mi) {
      int row = wm0 + mi * 16 + l15;
      af[mi] = cat16(*(const bf16x8*)&As[row][lh * 8],
                     *(const bf16x8*)&As[row][16 + lh * 8]);
    }
    #pragma unroll
    for (int ni = 0; ni < 4; ++ni) {
      int col = wn0 + ni * 16 + l15;
      bfrag[ni] = cat16(*(const bf16x8*)&Bs[col][lh * 8],
                        *(const bf16x8*)&Bs[col][16 + lh * 8]);
    }
    #pragma unroll
    for (int mi = 0; mi < 2; ++mi)
      #pragma unroll
      for (int ni = 0; ni < 4; ++ni)
        acc[mi][ni] = __builtin_amdgcn_wmma_f32_16x16x32_bf16(
            false, af[mi], false, bfrag[ni], (short)0, acc[mi][ni], false, false);
  }

  #pragma unroll
  for (int mi = 0; mi < 2; ++mi) {
    #pragma unroll
    for (int r = 0; r < 8; ++r) {
      int m = wm0 + mi * 16 + lh * 8 + r;
      int grow = r0 + m;
      if (grow < segLo || grow >= segHi) continue;
      #pragma unroll
      for (int ni = 0; ni < 4; ++ni) {
        int col = n0 + wn0 + ni * 16 + l15;
        float v = acc[mi][ni][r] + bias[col];
        hbuf[(size_t)grow * NO + col] = (__bf16)fmaxf(v, 0.f);
      }
    }
  }
}

// ---------------- GEMM layer 2: out[token] (=|+=) h @ W2 + b2 (+ cAdd) ----------------

__global__ __launch_bounds__(256) void k_gemm2(
    const __bf16* __restrict__ hbuf, const int* __restrict__ rowmap,
    const int* __restrict__ meta, const __bf16* __restrict__ BtBase,
    const float* __restrict__ biasBase, const float* __restrict__ cAdd,
    float* __restrict__ out, int mode) {
  constexpr int K = HIDDEN, NO = D_MODEL;
  __shared__ __bf16 As[BM][LDT];
  __shared__ __bf16 Bs[BN][LDT];
  __shared__ int rowTok[BM];

  int r0 = blockIdx.y * BM;
  int n0 = blockIdx.x * BN;
  int e = (mode == 1) ? (int)blockIdx.z : 0;
  int segLo = 0, segHi = NTOK;
  if (mode == 1) { int c0 = meta[0]; segLo = e ? c0 : 0; segHi = e ? NTOK : c0; }
  if (r0 >= segHi || r0 + BM <= segLo) return;

  const __bf16* Bt  = BtBase + (size_t)e * NO * K;
  const float* bias = biasBase + (size_t)e * NO;
  const float* cA   = cAdd + (size_t)e * NO;

  int tid = threadIdx.x, lane = tid & 31, wave = tid >> 5;
  int wm0 = (wave & 3) * 32, wn0 = (wave >> 2) * 64;
  int lh = lane >> 4, l15 = lane & 15;

  if (tid < BM) rowTok[tid] = rowmap[r0 + tid];
  __syncthreads();

  int lr = tid >> 3;
  int lc = (tid & 7) * 4;

  v8f acc[2][4];
  v8f vzero = {0.f,0.f,0.f,0.f,0.f,0.f,0.f,0.f};
  #pragma unroll
  for (int mi = 0; mi < 2; ++mi)
    #pragma unroll
    for (int ni = 0; ni < 4; ++ni) acc[mi][ni] = vzero;

  for (int kt = 0; kt < K; kt += BK) {
    bf16x4 av[4], bv[4];
    #pragma unroll
    for (int i = 0; i < 4; ++i) {
      int r = lr + i * 32;
      av[i] = *(const bf16x4*)(hbuf + (size_t)(r0 + r) * K + kt + lc);
      bv[i] = *(const bf16x4*)(Bt + (size_t)(n0 + r) * K + kt + lc);
    }
    __syncthreads();
    #pragma unroll
    for (int i = 0; i < 4; ++i) {
      int r = lr + i * 32;
      *(bf16x4*)&As[r][lc] = av[i];
      *(bf16x4*)&Bs[r][lc] = bv[i];
    }
    __syncthreads();

    bf16x16 af[2], bfrag[4];
    #pragma unroll
    for (int mi = 0; mi < 2; ++mi) {
      int row = wm0 + mi * 16 + l15;
      af[mi] = cat16(*(const bf16x8*)&As[row][lh * 8],
                     *(const bf16x8*)&As[row][16 + lh * 8]);
    }
    #pragma unroll
    for (int ni = 0; ni < 4; ++ni) {
      int col = wn0 + ni * 16 + l15;
      bfrag[ni] = cat16(*(const bf16x8*)&Bs[col][lh * 8],
                        *(const bf16x8*)&Bs[col][16 + lh * 8]);
    }
    #pragma unroll
    for (int mi = 0; mi < 2; ++mi)
      #pragma unroll
      for (int ni = 0; ni < 4; ++ni)
        acc[mi][ni] = __builtin_amdgcn_wmma_f32_16x16x32_bf16(
            false, af[mi], false, bfrag[ni], (short)0, acc[mi][ni], false, false);
  }

  #pragma unroll
  for (int mi = 0; mi < 2; ++mi) {
    #pragma unroll
    for (int r = 0; r < 8; ++r) {
      int m = wm0 + mi * 16 + lh * 8 + r;
      int grow = r0 + m;
      if (grow < segLo || grow >= segHi) continue;
      int tok = rowTok[m];
      #pragma unroll
      for (int ni = 0; ni < 4; ++ni) {
        int col = n0 + wn0 + ni * 16 + l15;
        float v = acc[mi][ni][r] + bias[col];
        float* op = out + (size_t)tok * NO + col;
        if (mode == 0) *op = v;
        else           *op = *op + v + cA[col];
      }
    }
  }
}

extern "C" void kernel_launch(void* const* d_in, const int* in_sizes, int n_in,
                              void* d_out, int out_size, void* d_ws, size_t ws_size,
                              hipStream_t stream) {
  const float* x   = (const float*)d_in[0];
  const float* gw  = (const float*)d_in[1];
  const float* w1  = (const float*)d_in[2];
  const float* b1  = (const float*)d_in[3];
  const float* w2  = (const float*)d_in[4];
  const float* b2  = (const float*)d_in[5];
  const float* sw1 = (const float*)d_in[6];
  const float* sb1 = (const float*)d_in[7];
  const float* sw2 = (const float*)d_in[8];
  const float* sb2 = (const float*)d_in[9];
  float* out = (float*)d_out;

  char* ws = (char*)d_ws;
  size_t off = 0;
  auto alloc = [&](size_t bytes) { size_t p = off; off += (bytes + 255) & ~(size_t)255; return p; };
  int*    meta   = (int*)(ws + alloc(2 * sizeof(int)));
  int*    rowmap = (int*)(ws + alloc((size_t)NTOK * sizeof(int)));
  float*  cAdd   = (float*)(ws + alloc((size_t)2 * D_MODEL * sizeof(float)));
  __bf16* sw1t   = (__bf16*)(ws + alloc((size_t)HIDDEN * D_MODEL * 2));
  __bf16* w1t    = (__bf16*)(ws + alloc((size_t)2 * HIDDEN * D_MODEL * 2));
  __bf16* sw2t   = (__bf16*)(ws + alloc((size_t)D_MODEL * HIDDEN * 2));
  __bf16* w2t    = (__bf16*)(ws + alloc((size_t)2 * D_MODEL * HIDDEN * 2));
  __bf16* hbuf   = (__bf16*)(ws + alloc((size_t)NTOK * HIDDEN * 2));

  k_init<<<1, 32, 0, stream>>>(meta);
  k_gate<<<NTOK / 8, 256, 0, stream>>>(x, gw, meta, rowmap);

  dim3 tb(32, 8);
  k_tconv<<<dim3(HIDDEN / 32, D_MODEL / 32), tb, 0, stream>>>(sw1, sw1t, D_MODEL, HIDDEN);
  k_tconv<<<dim3(HIDDEN / 32, D_MODEL / 32), tb, 0, stream>>>(w1, w1t, D_MODEL, HIDDEN);
  k_tconv<<<dim3(HIDDEN / 32, D_MODEL / 32), tb, 0, stream>>>(
      w1 + (size_t)D_MODEL * HIDDEN, w1t + (size_t)HIDDEN * D_MODEL, D_MODEL, HIDDEN);
  k_tconv<<<dim3(D_MODEL / 32, HIDDEN / 32), tb, 0, stream>>>(sw2, sw2t, HIDDEN, D_MODEL);
  k_tconv<<<dim3(D_MODEL / 32, HIDDEN / 32), tb, 0, stream>>>(w2, w2t, HIDDEN, D_MODEL);
  k_tconv<<<dim3(D_MODEL / 32, HIDDEN / 32), tb, 0, stream>>>(
      w2 + (size_t)HIDDEN * D_MODEL, w2t + (size_t)D_MODEL * HIDDEN, HIDDEN, D_MODEL);

  k_const<<<dim3(D_MODEL / 256, 2), 256, 0, stream>>>(b1, w2, b2, cAdd);

  // shared expert: assign into out
  k_gemm1<<<dim3(HIDDEN / BN, NTOK / BM, 1), 256, 0, stream>>>(x, rowmap, meta, sw1t, sb1, hbuf, 0);
  k_gemm2<<<dim3(D_MODEL / BN, NTOK / BM, 1), 256, 0, stream>>>(hbuf, rowmap, meta, sw2t, sb2, cAdd, out, 0);
  // routed experts (z = expert id, segmented by c0): accumulate + leaked constant
  k_gemm1<<<dim3(HIDDEN / BN, NTOK / BM, 2), 256, 0, stream>>>(x, rowmap, meta, w1t, b1, hbuf, 1);
  k_gemm2<<<dim3(D_MODEL / BN, NTOK / BM, 2), 256, 0, stream>>>(hbuf, rowmap, meta, w2t, b2, cAdd, out, 1);
}
